// FusionModel_11897059410618
// MI455X (gfx1250) — compile-verified, matrix operand order checked by reading, not run on hardware
//
#include <hip/hip_runtime.h>

typedef __attribute__((ext_vector_type(16))) __bf16 v16bf;
typedef __attribute__((ext_vector_type(8)))  __bf16 v8bf;
typedef __attribute__((ext_vector_type(8)))  float  v8f;

#define HFEAT 64

// ---------------- conversion / transpose helpers ----------------

__global__ void k_conv_bf16(const float* __restrict__ in, __bf16* __restrict__ out, long n) {
    long i = (long)blockIdx.x * blockDim.x + threadIdx.x;
    if (i < n) out[i] = (__bf16)in[i];
}

// in: [K x N] f32 row-major  ->  out: [N x K] bf16 row-major (B^T for WMMA)
__global__ void k_tconv_bf16(const float* __restrict__ in, __bf16* __restrict__ out, int K, int N) {
    int idx = blockIdx.x * blockDim.x + threadIdx.x;
    if (idx >= K * N) return;
    int k = idx / N, n = idx % N;
    out[(size_t)n * K + k] = (__bf16)in[idx];
}

// ---------------- WMMA GEMM: C[MxN] = A[MxK] * Bt[NxK]^T (+bias)(+relu) ----------------
// wave32; one wave -> one 16x64 output slab (4 WMMA tiles). A fragment is loaded once
// per K-step and reused across 4 back-to-back v_wmma ops (4x less A traffic than
// tile-per-wave). K multiple of 32; M,N multiples of 16; gridDim.y covers N in 64-col slabs.

__global__ void k_gemm_wmma_bf16(const __bf16* __restrict__ A,
                                 const __bf16* __restrict__ Bt,
                                 const float* __restrict__ bias,
                                 float* __restrict__ C,
                                 int M, int K, int N, int relu) {
    const int wave = threadIdx.x >> 5;
    const int lane = threadIdx.x & 31;
    const int m0 = (blockIdx.x * (blockDim.x >> 5) + wave) * 16;
    if (m0 >= M) return;                         // wave-uniform: EXEC stays all-ones
    const int ncol0 = blockIdx.y * 64;
    const int ntiles = min(4, (N - ncol0) >> 4); // 4 for all shapes used here

    const int lrow = lane & 15;   // A row / Bt row within tile
    const int hi   = lane >> 4;   // K-half select per ISA 16-bit operand layout

    const __bf16* arow = A + (size_t)(m0 + lrow) * K + hi * 8;
    const __bf16* brow[4];
#pragma unroll
    for (int t = 0; t < 4; ++t)
        brow[t] = Bt + (size_t)(ncol0 + t * 16 + lrow) * K + hi * 8;

    v8f acc0 = {}, acc1 = {}, acc2 = {}, acc3 = {};
    for (int k = 0; k < K; k += 32) {
        v8bf alo = *(const v8bf*)(arow + k);        // K = k+hi*8 .. +7
        v8bf ahi = *(const v8bf*)(arow + k + 16);   // K = k+16+hi*8 .. +7
        v16bf a = __builtin_shufflevector(alo, ahi, 0,1,2,3,4,5,6,7,8,9,10,11,12,13,14,15);

        v16bf b0, b1, b2, b3;
        {
            v8bf lo = *(const v8bf*)(brow[0] + k), hv = *(const v8bf*)(brow[0] + k + 16);
            b0 = __builtin_shufflevector(lo, hv, 0,1,2,3,4,5,6,7,8,9,10,11,12,13,14,15);
            lo = *(const v8bf*)(brow[1] + k); hv = *(const v8bf*)(brow[1] + k + 16);
            b1 = __builtin_shufflevector(lo, hv, 0,1,2,3,4,5,6,7,8,9,10,11,12,13,14,15);
            lo = *(const v8bf*)(brow[2] + k); hv = *(const v8bf*)(brow[2] + k + 16);
            b2 = __builtin_shufflevector(lo, hv, 0,1,2,3,4,5,6,7,8,9,10,11,12,13,14,15);
            lo = *(const v8bf*)(brow[3] + k); hv = *(const v8bf*)(brow[3] + k + 16);
            b3 = __builtin_shufflevector(lo, hv, 0,1,2,3,4,5,6,7,8,9,10,11,12,13,14,15);
        }
        acc0 = __builtin_amdgcn_wmma_f32_16x16x32_bf16(false, a, false, b0, (short)0, acc0, false, false);
        acc1 = __builtin_amdgcn_wmma_f32_16x16x32_bf16(false, a, false, b1, (short)0, acc1, false, false);
        acc2 = __builtin_amdgcn_wmma_f32_16x16x32_bf16(false, a, false, b2, (short)0, acc2, false, false);
        acc3 = __builtin_amdgcn_wmma_f32_16x16x32_bf16(false, a, false, b3, (short)0, acc3, false, false);
    }

    // C/D layout: lane = (hi, lrow); VGPR r -> M = r + 8*hi, N = tile col + lrow
    const size_t rowbase = (size_t)(m0 + hi * 8) * N;
    v8f accs[4] = {acc0, acc1, acc2, acc3};
#pragma unroll
    for (int t = 0; t < 4; ++t) {
        if (t >= ntiles) break;
        const int col = ncol0 + t * 16 + lrow;
        const float bv = bias ? bias[col] : 0.0f;
#pragma unroll
        for (int r = 0; r < 8; ++r) {
            float v = accs[t][r] + bv;
            if (relu) v = fmaxf(v, 0.0f);
            C[rowbase + (size_t)r * N + col] = v;
        }
    }
}

// ---------------- graph degree / normalization ----------------

__global__ void k_degree(const int* __restrict__ dst, float* __restrict__ deg, int E) {
    int e = blockIdx.x * blockDim.x + threadIdx.x;
    if (e < E) atomicAdd(&deg[dst[e]], 1.0f);
}

__global__ void k_dis(float* __restrict__ deg, int N) {
    int i = blockIdx.x * blockDim.x + threadIdx.x;
    if (i < N) deg[i] = rsqrtf(deg[i] + 1.0f);      // in-place: deg -> dis
}

// ---------------- edge scatter: agg[dst] += h[src] * (dis[src]*dis[dst]) ----------------
// 16 lanes per edge, float4 per lane -> coalesced 256B row reads (L2-resident).

__global__ void k_edge_agg(const float* __restrict__ h,
                           const int* __restrict__ src, const int* __restrict__ dst,
                           const float* __restrict__ dis,
                           float* __restrict__ agg, int E) {
    long idx = (long)blockIdx.x * blockDim.x + threadIdx.x;
    int e = (int)(idx >> 4);
    if (e >= E) return;
    int f = ((int)idx & 15) * 4;
    int s = src[e], d = dst[e];
    if ((idx & 15) == 0 && e + 256 < E) {
        __builtin_prefetch(&h[(size_t)src[e + 256] * HFEAT], 0, 0);  // global_prefetch_b8
    }
    float nrm = dis[s] * dis[d];
    float4 hv = *(const float4*)(h + (size_t)s * HFEAT + f);
    float* o = agg + (size_t)d * HFEAT + f;
    atomicAdd(o + 0, hv.x * nrm);
    atomicAdd(o + 1, hv.y * nrm);
    atomicAdd(o + 2, hv.z * nrm);
    atomicAdd(o + 3, hv.w * nrm);
}

// agg += h * dis^2 (self loop) + bias
__global__ void k_selfloop_bias(float* __restrict__ agg, const float* __restrict__ h,
                                const float* __restrict__ dis, const float* __restrict__ b,
                                int N) {
    long idx = (long)blockIdx.x * blockDim.x + threadIdx.x;
    if (idx >= (long)N * HFEAT) return;
    int i = (int)(idx >> 6), f = (int)idx & 63;
    float di = dis[i];
    agg[idx] += h[idx] * di * di + b[f];
}

// ---------------- batch norm ----------------

__global__ void k_bn_stats(const float* __restrict__ x, float* __restrict__ stats,
                           int N, int rows_per_block) {
    int f = threadIdx.x;                       // 64 threads: one feature per thread
    int r0 = blockIdx.x * rows_per_block;
    int r1 = min(r0 + rows_per_block, N);
    float s = 0.0f, q = 0.0f;
    for (int r = r0; r < r1; ++r) {
        float v = x[(size_t)r * HFEAT + f];
        s += v; q += v * v;
    }
    atomicAdd(&stats[f], s);
    atomicAdd(&stats[64 + f], q);
}

__global__ void k_bn_final(float* __restrict__ stats, const float* __restrict__ g,
                           const float* __restrict__ be, int N) {
    int f = threadIdx.x;
    float inv_n = 1.0f / (float)N;
    float mu  = stats[f] * inv_n;
    float var = stats[64 + f] * inv_n - mu * mu;
    float sc  = g[f] * rsqrtf(var + 1e-5f);
    stats[128 + f] = sc;
    stats[192 + f] = be[f] - mu * sc;
}

// out = relu(x*scale + shift [+ res])
__global__ void k_bn_apply(const float* __restrict__ x, const float* __restrict__ stats,
                           const float* __restrict__ res, float* __restrict__ out, int N) {
    long idx = (long)blockIdx.x * blockDim.x + threadIdx.x;
    if (idx >= (long)N * HFEAT) return;
    int f = (int)idx & 63;
    float v = x[idx] * stats[128 + f] + stats[192 + f];
    if (res) v += res[idx];
    out[idx] = fmaxf(v, 0.0f);
}

// ---------------- mean pooling over graphs ----------------

__global__ void k_pool(const float* __restrict__ act, const int* __restrict__ batch,
                       float* __restrict__ pool, float* __restrict__ cnt, int N) {
    long idx = (long)blockIdx.x * blockDim.x + threadIdx.x;
    if (idx >= (long)N * HFEAT) return;
    int i = (int)(idx >> 6), f = (int)idx & 63;
    int g = batch[i];
    atomicAdd(&pool[(size_t)g * HFEAT + f], act[idx]);
    if (f == 0) atomicAdd(&cnt[g], 1.0f);
}

__global__ void k_pool_div(const float* __restrict__ pool, const float* __restrict__ cnt,
                           float* __restrict__ emb, int G) {
    int idx = blockIdx.x * blockDim.x + threadIdx.x;
    if (idx >= G * HFEAT) return;
    int g = idx >> 6;
    emb[idx] = pool[idx] / fmaxf(cnt[g], 1.0f);
}

// ---------------- fusion head: out[g] = [emb|f2] . Wfu + bfu ----------------

__global__ void k_fusion(const float* __restrict__ emb, const float* __restrict__ f2,
                         const float* __restrict__ Wfu, const float* __restrict__ bfu,
                         float* __restrict__ out, int G) {
    int g = blockIdx.x * blockDim.x + threadIdx.x;
    if (g >= G) return;
    float s = bfu[0];
#pragma unroll
    for (int h = 0; h < HFEAT; ++h) s += emb[(size_t)g * HFEAT + h] * Wfu[h];
#pragma unroll
    for (int h = 0; h < HFEAT; ++h) s += f2[(size_t)g * HFEAT + h] * Wfu[HFEAT + h];
    out[g] = s;
}

// ---------------- host orchestration ----------------

extern "C" void kernel_launch(void* const* d_in, const int* in_sizes, int n_in,
                              void* d_out, int out_size, void* d_ws, size_t ws_size,
                              hipStream_t stream) {
    const float* x     = (const float*)d_in[0];
    const int*   ei    = (const int*)d_in[1];
    const int*   batch = (const int*)d_in[2];
    const float* sigma = (const float*)d_in[3];
    const float* W1 = (const float*)d_in[4],  *b1 = (const float*)d_in[5];
    const float* g1 = (const float*)d_in[6],  *be1 = (const float*)d_in[7];
    const float* W2 = (const float*)d_in[8],  *b2 = (const float*)d_in[9];
    const float* g2 = (const float*)d_in[10], *be2 = (const float*)d_in[11];
    const float* W3 = (const float*)d_in[12], *b3 = (const float*)d_in[13];
    const float* g3 = (const float*)d_in[14], *be3 = (const float*)d_in[15];
    const float* Wf1 = (const float*)d_in[16], *bf1 = (const float*)d_in[17];
    const float* Wf2 = (const float*)d_in[18], *bf2 = (const float*)d_in[19];
    const float* Wfu = (const float*)d_in[20], *bfu = (const float*)d_in[21];
    float* out = (float*)d_out;

    const int Fin = 128, H = 64, S = 64;
    const int N = in_sizes[0] / Fin;
    const int E = in_sizes[1] / 2;
    const int G = in_sizes[3] / S;
    const int* src = ei;
    const int* dst = ei + E;

    // ---- workspace carve-up (256B aligned) ----
    char* base = (char*)d_ws;
    size_t off = 0;
    auto take = [&](size_t bytes) -> char* {
        char* p = base + off;
        off += (bytes + 255) & ~(size_t)255;
        return p;
    };
    __bf16* xb   = (__bf16*)take((size_t)N * Fin * 2);
    __bf16* hb   = (__bf16*)take((size_t)N * H * 2);
    float*  hg   = (float*)take((size_t)N * H * 4);     // GEMM output h = act @ W
    float*  agg  = (float*)take((size_t)N * H * 4);     // scatter accumulator
    float*  actA = (float*)take((size_t)N * H * 4);
    float*  actB = (float*)take((size_t)N * H * 4);
    float*  dis  = (float*)take((size_t)N * 4);
    float*  stats= (float*)take(256 * 4);
    float*  pool = (float*)take((size_t)G * H * 4);
    float*  cnt  = (float*)take((size_t)G * 4);
    float*  emb  = (float*)take((size_t)G * H * 4);
    __bf16* sigb = (__bf16*)take((size_t)G * S * 2);
    float*  f1   = (float*)take((size_t)G * 2 * H * 4);
    __bf16* f1b  = (__bf16*)take((size_t)G * 2 * H * 2);
    float*  f2   = (float*)take((size_t)G * H * 4);
    __bf16* w1t  = (__bf16*)take((size_t)H * Fin * 2);
    __bf16* w2t  = (__bf16*)take((size_t)H * H * 2);
    __bf16* w3t  = (__bf16*)take((size_t)H * H * 2);
    __bf16* wf1t = (__bf16*)take((size_t)(2 * H) * S * 2);
    __bf16* wf2t = (__bf16*)take((size_t)H * (2 * H) * 2);
    (void)ws_size; (void)n_in;

    const int TB = 256;
    auto blocks = [&](long n) { return dim3((unsigned)((n + TB - 1) / TB)); };
    // GEMM launch geometry: 8 waves/block, 1 M-tile (16 rows) per wave, 64 cols per wave
    auto gemmGrid = [&](int M, int Ncols) {
        return dim3((unsigned)(((M / 16) + 7) / 8), (unsigned)((Ncols + 63) / 64));
    };
    dim3 gemmBlk(256);

    // ---- prepare bf16 operands ----
    k_conv_bf16<<<blocks((long)N * Fin), TB, 0, stream>>>(x, xb, (long)N * Fin);
    k_conv_bf16<<<blocks((long)G * S), TB, 0, stream>>>(sigma, sigb, (long)G * S);
    k_tconv_bf16<<<blocks((long)Fin * H), TB, 0, stream>>>(W1, w1t, Fin, H);
    k_tconv_bf16<<<blocks((long)H * H), TB, 0, stream>>>(W2, w2t, H, H);
    k_tconv_bf16<<<blocks((long)H * H), TB, 0, stream>>>(W3, w3t, H, H);
    k_tconv_bf16<<<blocks((long)S * 2 * H), TB, 0, stream>>>(Wf1, wf1t, S, 2 * H);
    k_tconv_bf16<<<blocks((long)2 * H * H), TB, 0, stream>>>(Wf2, wf2t, 2 * H, H);

    // ---- degree / symmetric normalization ----
    hipMemsetAsync(dis, 0, (size_t)N * 4, stream);
    k_degree<<<blocks(E), TB, 0, stream>>>(dst, dis, E);
    k_dis<<<blocks(N), TB, 0, stream>>>(dis, N);

    const int rpb = (N + 255) / 256;
    const long NH = (long)N * H;

    // =============== GCN layer 1 ===============
    k_gemm_wmma_bf16<<<gemmGrid(N, H), gemmBlk, 0, stream>>>(xb, w1t, nullptr, hg, N, Fin, H, 0);
    hipMemsetAsync(agg, 0, (size_t)N * H * 4, stream);
    k_edge_agg<<<blocks((long)E * 16), TB, 0, stream>>>(hg, src, dst, dis, agg, E);
    k_selfloop_bias<<<blocks(NH), TB, 0, stream>>>(agg, hg, dis, b1, N);
    hipMemsetAsync(stats, 0, 256 * 4, stream);
    k_bn_stats<<<dim3(256), dim3(64), 0, stream>>>(agg, stats, N, rpb);
    k_bn_final<<<dim3(1), dim3(64), 0, stream>>>(stats, g1, be1, N);
    k_bn_apply<<<blocks(NH), TB, 0, stream>>>(agg, stats, nullptr, actA, N);
    k_conv_bf16<<<blocks(NH), TB, 0, stream>>>(actA, hb, NH);

    // =============== GCN layer 2 (+residual actA) ===============
    k_gemm_wmma_bf16<<<gemmGrid(N, H), gemmBlk, 0, stream>>>(hb, w2t, nullptr, hg, N, H, H, 0);
    hipMemsetAsync(agg, 0, (size_t)N * H * 4, stream);
    k_edge_agg<<<blocks((long)E * 16), TB, 0, stream>>>(hg, src, dst, dis, agg, E);
    k_selfloop_bias<<<blocks(NH), TB, 0, stream>>>(agg, hg, dis, b2, N);
    hipMemsetAsync(stats, 0, 256 * 4, stream);
    k_bn_stats<<<dim3(256), dim3(64), 0, stream>>>(agg, stats, N, rpb);
    k_bn_final<<<dim3(1), dim3(64), 0, stream>>>(stats, g2, be2, N);
    k_bn_apply<<<blocks(NH), TB, 0, stream>>>(agg, stats, actA, actB, N);
    k_conv_bf16<<<blocks(NH), TB, 0, stream>>>(actB, hb, NH);

    // =============== GCN layer 3 ===============
    k_gemm_wmma_bf16<<<gemmGrid(N, H), gemmBlk, 0, stream>>>(hb, w3t, nullptr, hg, N, H, H, 0);
    hipMemsetAsync(agg, 0, (size_t)N * H * 4, stream);
    k_edge_agg<<<blocks((long)E * 16), TB, 0, stream>>>(hg, src, dst, dis, agg, E);
    k_selfloop_bias<<<blocks(NH), TB, 0, stream>>>(agg, hg, dis, b3, N);
    hipMemsetAsync(stats, 0, 256 * 4, stream);
    k_bn_stats<<<dim3(256), dim3(64), 0, stream>>>(agg, stats, N, rpb);
    k_bn_final<<<dim3(1), dim3(64), 0, stream>>>(stats, g3, be3, N);
    k_bn_apply<<<blocks(NH), TB, 0, stream>>>(agg, stats, nullptr, actA, N);

    // =============== mean pool ===============
    hipMemsetAsync(pool, 0, (size_t)G * H * 4, stream);
    hipMemsetAsync(cnt, 0, (size_t)G * 4, stream);
    k_pool<<<blocks(NH), TB, 0, stream>>>(actA, batch, pool, cnt, N);
    k_pool_div<<<blocks((long)G * H), TB, 0, stream>>>(pool, cnt, emb, G);

    // =============== feed-forward branch (WMMA) ===============
    k_gemm_wmma_bf16<<<gemmGrid(G, 2 * H), gemmBlk, 0, stream>>>(sigb, wf1t, bf1, f1, G, S, 2 * H, 1);
    k_conv_bf16<<<blocks((long)G * 2 * H), TB, 0, stream>>>(f1, f1b, (long)G * 2 * H);
    k_gemm_wmma_bf16<<<gemmGrid(G, H), gemmBlk, 0, stream>>>(f1b, wf2t, bf2, f2, G, 2 * H, H, 1);

    // =============== fusion head ===============
    k_fusion<<<blocks(G), TB, 0, stream>>>(emb, f2, Wfu, bfu, out, G);
}
